// JointLatentLanguageModelPrefixSearch_64141041599166
// MI455X (gfx1250) — compile-verified, matrix operand order checked by reading, not run on hardware
//
#include <hip/hip_runtime.h>
#include <hip/hip_bf16.h>

#define VSZ   8000
#define VP1   8001
#define DM    1024
#define TST   24
#define NB    16
#define KW    16
#define LDC   8016              // padded logits columns = 501*16
#define NTI   (LDC / 16)        // 501 N-tiles
#define MBLK  4                 // M-tiles per wave (B-fragment reuse)
#define KCH   (DM / 32)         // 32 K-chunks

typedef __attribute__((ext_vector_type(16))) __bf16 v16bf;
typedef __attribute__((ext_vector_type(8)))  float  v8f;

union FragU { uint4 u[2]; v16bf v; };

__device__ __forceinline__ unsigned short f2bf(float f) {
  unsigned int u = __float_as_uint(f);
  u += 0x7FFFu + ((u >> 16) & 1u);     // round-to-nearest-even
  return (unsigned short)(u >> 16);
}

// ---------------------------------------------------------------------------
// One-time: out_proj (DM x VP1, f32, row-major) -> Wt (LDC x DM, bf16),
// Wt row n = column n of out_proj (zero-padded rows for n >= VP1).
// ---------------------------------------------------------------------------
__global__ void k_transpose_w(const float* __restrict__ W,
                              unsigned short* __restrict__ Wt) {
  int k = blockIdx.x;                              // 0..DM-1
  for (int n = threadIdx.x; n < LDC; n += blockDim.x) {
    float v = (n < VP1) ? W[(size_t)k * VP1 + n] : 0.0f;
    Wt[(size_t)n * DM + k] = f2bf(v);
  }
}

// ---------------------------------------------------------------------------
// State init: probs=1, y=V, cond=0, lens=0, prefix=1
// ---------------------------------------------------------------------------
__global__ void k_init_state(float* __restrict__ probs, int* __restrict__ y,
                             int* __restrict__ cond, int* __restrict__ lens,
                             int* __restrict__ pref) {
  int tid = blockIdx.x * blockDim.x + threadIdx.x;
  int nth = gridDim.x * blockDim.x;
  if (tid < NB * KW) { probs[tid] = 1.0f; lens[tid] = 0; }
  if (tid < NB * KW * KW) pref[tid] = 1;
  for (int i = tid; i < TST * NB * KW; i += nth) { y[i] = VSZ; cond[i] = 0; }
}

// ---------------------------------------------------------------------------
// h = token_emb[prev_tok] + time_emb[t], written as bf16 rows of H (M x DM)
// ---------------------------------------------------------------------------
__global__ void k_embed(const float* __restrict__ tokE,
                        const float* __restrict__ timE,
                        const int* __restrict__ yPrev,
                        unsigned short* __restrict__ H, int t, int Kp) {
  int row = blockIdx.x;                            // 0..NB*Kp-1
  int n = row / Kp, i = row % Kp;
  int prev = (t > 0) ? yPrev[(t - 1) * NB * KW + n * KW + i] : VSZ;
  const float* te = tokE + (size_t)prev * DM;
  const float* ti = timE + (size_t)t * DM;
  unsigned short* h = H + (size_t)row * DM;
  for (int k = threadIdx.x; k < DM; k += blockDim.x)
    h[k] = f2bf(te[k] + ti[k]);
}

// ---------------------------------------------------------------------------
// C(M x LDC, f32) = H(M x DM, bf16) * Wt^T  via v_wmma_f32_16x16x32_bf16.
// Each wave computes an MBLK*16 x 16 strip: one B fragment feeds MBLK WMMAs
// per K-chunk (4x less B traffic from L2 -- the dominant stream, since A is
// 512KB/L2-resident while B is 16.4MB re-read per M-row).
// Manual 2x K-unroll with ping-pong fragment sets: loads for chunk kc+1 are
// in flight while WMMAs consume chunk kc, with zero register copies.
// Fragment layout (16-bit A/B): lane L -> row/col (L&15),
// K elems [(L>>4)*8 .. +7] and [+16 .. +23]  => two uint4 loads.
// ---------------------------------------------------------------------------
__global__ void k_gemm(const unsigned short* __restrict__ H,
                       const unsigned short* __restrict__ Wt,
                       float* __restrict__ C, int M) {
  int wave = (blockIdx.x * blockDim.x + threadIdx.x) >> 5;
  int lane = threadIdx.x & 31;
  int Mt   = M >> 4;
  int Mrow = (Mt + MBLK - 1) / MBLK;               // wave rows of MBLK tiles
  if (wave >= Mrow * NTI) return;                  // wave-uniform exit
  int mr = wave / NTI, nt = wave % NTI;
  int half = lane >> 4;
  int l15  = lane & 15;

  const unsigned short* bRow = Wt + ((size_t)(nt * 16 + l15)) * DM + half * 8;
  const unsigned short* aRow[MBLK];
#pragma unroll
  for (int m = 0; m < MBLK; ++m) {
    int mt = mr * MBLK + m;
    int r  = (mt < Mt) ? (mt * 16 + l15) : l15;    // clamp (store is skipped)
    aRow[m] = H + (size_t)r * DM + half * 8;
  }

  v8f acc[MBLK];
#pragma unroll
  for (int m = 0; m < MBLK; ++m) acc[m] = (v8f){};

  FragU b0, b1, a0[MBLK], a1[MBLK];
  // prologue: chunk 0 -> set 0
  b0.u[0] = *(const uint4*)(bRow);
  b0.u[1] = *(const uint4*)(bRow + 16);
#pragma unroll
  for (int m = 0; m < MBLK; ++m) {
    a0[m].u[0] = *(const uint4*)(aRow[m]);
    a0[m].u[1] = *(const uint4*)(aRow[m] + 16);
  }

  for (int kc = 0; kc < KCH; kc += 2) {            // KCH is even
    // issue loads for chunk kc+1 -> set 1 (independent of set-0 WMMAs)
    b1.u[0] = *(const uint4*)(bRow + (kc + 1) * 32);
    b1.u[1] = *(const uint4*)(bRow + (kc + 1) * 32 + 16);
#pragma unroll
    for (int m = 0; m < MBLK; ++m) {
      a1[m].u[0] = *(const uint4*)(aRow[m] + (kc + 1) * 32);
      a1[m].u[1] = *(const uint4*)(aRow[m] + (kc + 1) * 32 + 16);
    }
    __builtin_prefetch(bRow + (kc + 2) * 32, 0, 3);   // global_prefetch_b8
    // compute chunk kc from set 0
#pragma unroll
    for (int m = 0; m < MBLK; ++m)
      acc[m] = __builtin_amdgcn_wmma_f32_16x16x32_bf16(
          false, a0[m].v, false, b0.v, (short)0, acc[m], false, false);
    // issue loads for chunk kc+2 -> set 0 (independent of set-1 WMMAs)
    if (kc + 2 < KCH) {
      b0.u[0] = *(const uint4*)(bRow + (kc + 2) * 32);
      b0.u[1] = *(const uint4*)(bRow + (kc + 2) * 32 + 16);
#pragma unroll
      for (int m = 0; m < MBLK; ++m) {
        a0[m].u[0] = *(const uint4*)(aRow[m] + (kc + 2) * 32);
        a0[m].u[1] = *(const uint4*)(aRow[m] + (kc + 2) * 32 + 16);
      }
    }
    // compute chunk kc+1 from set 1
#pragma unroll
    for (int m = 0; m < MBLK; ++m)
      acc[m] = __builtin_amdgcn_wmma_f32_16x16x32_bf16(
          false, a1[m].v, false, b1.v, (short)0, acc[m], false, false);
  }

  // D layout: element (M = r + 8*half, N = l15) in acc[m][r]
#pragma unroll
  for (int m = 0; m < MBLK; ++m) {
    int mt = mr * MBLK + m;
    if (mt < Mt) {                                 // wave-uniform guard
      float* cp = C + ((size_t)(mt * 16 + half * 8)) * LDC + nt * 16 + l15;
#pragma unroll
      for (int r = 0; r < 8; ++r) cp[(size_t)r * LDC] = acc[m][r];
    }
  }
}

// ---------------------------------------------------------------------------
// Row-wise softmax over VP1 entries (pads ignored), in place.
// ---------------------------------------------------------------------------
__global__ void k_softmax(float* __restrict__ P) {
  float* p = P + (size_t)blockIdx.x * LDC;
  __shared__ float red[256];
  int tid = threadIdx.x;
  float mx = -__builtin_inff();
  for (int v = tid; v < VP1; v += 256) mx = fmaxf(mx, p[v]);
  red[tid] = mx; __syncthreads();
  for (int s = 128; s > 0; s >>= 1) {
    if (tid < s) red[tid] = fmaxf(red[tid], red[tid + s]);
    __syncthreads();
  }
  mx = red[0]; __syncthreads();
  float sum = 0.0f;
  for (int v = tid; v < VP1; v += 256) sum += __expf(p[v] - mx);
  red[tid] = sum; __syncthreads();
  for (int s = 128; s > 0; s >>= 1) {
    if (tid < s) red[tid] += red[tid + s];
    __syncthreads();
  }
  float inv = 1.0f / red[0]; __syncthreads();
  for (int v = tid; v < VP1; v += 256) p[v] = __expf(p[v] - mx) * inv;
}

// ---------------------------------------------------------------------------
// Per batch element n: exact-prefix merge into nonext, mask duplicates,
// top-16 over Kp*V + Kp candidates (single pass, per-thread register top-16
// + cross-thread merge), then full beam-state update.
// ---------------------------------------------------------------------------
__global__ void k_topk_update(
    float* __restrict__ P,                          // (M x LDC) probs, masked in place
    const float* __restrict__ probsPrev, const int* __restrict__ yPrev,
    const int* __restrict__ condPrev, const int* __restrict__ lensPrev,
    const int* __restrict__ prefPrev,
    float* __restrict__ probsNext, int* __restrict__ yNext,
    int* __restrict__ condNext, int* __restrict__ lensNext,
    int* __restrict__ prefNext, int t, int Kp) {
  int n = blockIdx.x;
  int tid = threadIdx.x;

  __shared__ float s_pp[KW];
  __shared__ int   s_len[KW];
  __shared__ float s_non[KW];
  __shared__ int   s_pref[KW][KW];
  __shared__ int   s_tok[KW][KW];
  __shared__ float s_rv[256];
  __shared__ int   s_ri[256];
  __shared__ int   s_selI[KW];
  __shared__ float s_selV[KW];
  __shared__ int   s_src[KW], s_ext[KW], s_nx[KW], s_nl[KW];

  if (tid < Kp) { s_pp[tid] = probsPrev[n * KW + tid];
                  s_len[tid] = lensPrev[n * KW + tid]; }
  __syncthreads();
  if (tid < Kp * Kp) {
    int i = tid / Kp, j = tid % Kp;
    s_pref[i][j] = prefPrev[n * KW * KW + i * KW + j];
    s_tok[i][j]  = condPrev[s_len[i] * NB * KW + n * KW + j];
  }
  __syncthreads();
  if (tid < Kp) {                                   // nonext + exact-merge mass
    int j = tid;
    float nx = P[((size_t)(n * Kp + j)) * LDC + VSZ] * s_pp[j];
    for (int i = 0; i < Kp; ++i) {
      bool exact = (s_len[i] + 1 == s_len[j]) && (s_pref[i][j] != 0);
      if (exact) nx += P[((size_t)(n * Kp + i)) * LDC + s_tok[i][j]] * s_pp[i];
    }
    s_non[j] = nx;
  }
  __syncthreads();
  if (tid < Kp * Kp) {                              // mask exact duplicates
    int i = tid / Kp, j = tid % Kp;
    bool exact = (s_len[i] + 1 == s_len[j]) && (s_pref[i][j] != 0);
    if (exact) P[((size_t)(n * Kp + i)) * LDC + s_tok[i][j]] = -__builtin_inff();
  }
  __syncthreads();

  // ---- per-thread register top-16 (value desc, index asc on ties) ----
  const int ncand = Kp * VSZ + Kp;
  float lv[KW]; int li[KW];
#pragma unroll
  for (int q = 0; q < KW; ++q) { lv[q] = -__builtin_inff(); li[q] = 0x7FFFFFFF; }
  for (int c = tid; c < ncand; c += 256) {
    float v;
    if (c < Kp * VSZ) {
      int i = c / VSZ;
      v = P[((size_t)(n * Kp + i)) * LDC + (c - i * VSZ)] * s_pp[i];
    } else {
      v = s_non[c - Kp * VSZ];
    }
    if (v > lv[KW - 1] || (v == lv[KW - 1] && c < li[KW - 1])) {
      int pos = KW - 1;
#pragma unroll
      for (int q = KW - 2; q >= 0; --q)
        if (v > lv[q] || (v == lv[q] && c < li[q])) pos = q;
#pragma unroll
      for (int q = KW - 1; q > 0; --q)
        if (q > pos) { lv[q] = lv[q - 1]; li[q] = li[q - 1]; }
      lv[pos] = v; li[pos] = c;
    }
  }

  // ---- merge: 16 rounds of block argmax over per-thread sorted heads ----
  int ptr = 0;
  for (int k = 0; k < KW; ++k) {
    float hv = (ptr < KW) ? lv[ptr] : -__builtin_inff();
    int   hi = (ptr < KW) ? li[ptr] : 0x7FFFFFFF;
    s_rv[tid] = hv; s_ri[tid] = hi;
    __syncthreads();
    for (int s = 128; s > 0; s >>= 1) {
      if (tid < s) {
        float ov = s_rv[tid + s]; int oi = s_ri[tid + s];
        if (ov > s_rv[tid] || (ov == s_rv[tid] && oi < s_ri[tid])) {
          s_rv[tid] = ov; s_ri[tid] = oi;
        }
      }
      __syncthreads();
    }
    if (tid == 0) { s_selI[k] = s_ri[0]; s_selV[k] = s_rv[0]; }
    __syncthreads();
    if (ptr < KW && li[ptr] == s_selI[k]) ++ptr;    // winner advances
    __syncthreads();
  }

  // ---- decode selections ----
  if (tid < KW) {
    int c = s_selI[tid];
    int nx  = (c >= Kp * VSZ) ? 1 : 0;
    int src = nx ? (c - Kp * VSZ) : (c / VSZ);
    int ext = c % VSZ;                              // matches reference (idx % V)
    s_src[tid] = src; s_ext[tid] = ext; s_nx[tid] = nx;
    s_nl[tid] = s_len[src] + (nx ? 0 : 1);
    lensNext[n * KW + tid]  = s_nl[tid];
    probsNext[n * KW + tid] = s_selV[tid] / s_selV[0];   // NORMALIZE (max = first)
  }
  __syncthreads();

  // ---- gather sequences / conditioned tokens for new beams ----
  for (int idx = tid; idx < TST * KW; idx += 256) {
    int tt = idx / KW, k = idx % KW;
    int src = s_src[k];
    int yv = yPrev[tt * NB * KW + n * KW + src];
    if (tt == t) yv = s_nx[k] ? VSZ : s_ext[k];
    yNext[tt * NB * KW + n * KW + k] = yv;
    int cv = condPrev[tt * NB * KW + n * KW + src];
    if (tt == s_len[src]) cv = s_ext[k];            // set at prev len of src
    condNext[tt * NB * KW + n * KW + k] = cv;
  }
  __syncthreads();

  // ---- next prefix relation (needs condNext written above) ----
  if (tid < KW * KW) {
    int a = tid / KW, b = tid % KW;
    int pp  = s_pref[s_src[a]][s_src[b]];
    int leq = (s_nl[a] <= s_nl[b]) ? 1 : 0;
    int lm1 = s_nl[a] - 1; if (lm1 < 0) lm1 = 0;
    int tomatch = condNext[lm1 * NB * KW + n * KW + b];
    int em  = (tomatch == s_ext[a]) ? 1 : 0;
    prefNext[n * KW * KW + a * KW + b] = (pp && leq && (s_nx[a] || em)) ? 1 : 0;
  }
}

// ---------------------------------------------------------------------------
// Pack outputs: (y_cond int[T*N*K], lens int[N*K], probs float[N*K])
// ---------------------------------------------------------------------------
__global__ void k_finalize(const int* __restrict__ cond,
                           const int* __restrict__ lens,
                           const float* __restrict__ probs, void* out) {
  int idx = blockIdx.x * blockDim.x + threadIdx.x;
  const int S1 = TST * NB * KW;
  const int S2 = S1 + NB * KW;
  const int S3 = S2 + NB * KW;
  int*   oi = (int*)out;
  float* of = (float*)out;
  if (idx < S1)       oi[idx] = cond[idx];
  else if (idx < S2)  oi[idx] = lens[idx - S1];
  else if (idx < S3)  of[idx] = probs[idx - S2];
}

// ---------------------------------------------------------------------------
extern "C" void kernel_launch(void* const* d_in, const int* in_sizes, int n_in,
                              void* d_out, int out_size, void* d_ws, size_t ws_size,
                              hipStream_t stream) {
  (void)in_sizes; (void)n_in; (void)out_size; (void)ws_size;
  // setup_inputs order: N, T, token_emb, time_emb, out_proj
  const float* tokE = (const float*)d_in[2];
  const float* timE = (const float*)d_in[3];
  const float* W    = (const float*)d_in[4];

  char* ws = (char*)d_ws;
  size_t off = 0;
  auto alloc = [&](size_t bytes) -> void* {
    void* p = ws + off;
    off = (off + bytes + 255) & ~(size_t)255;
    return p;
  };

  unsigned short* Wt = (unsigned short*)alloc((size_t)LDC * DM * 2);
  unsigned short* H  = (unsigned short*)alloc((size_t)NB * KW * DM * 2);
  float*          P  = (float*)alloc((size_t)NB * KW * LDC * 4);

  float* probsA = (float*)alloc(NB * KW * 4);
  float* probsB = (float*)alloc(NB * KW * 4);
  int* yA    = (int*)alloc(TST * NB * KW * 4);
  int* yB    = (int*)alloc(TST * NB * KW * 4);
  int* condA = (int*)alloc(TST * NB * KW * 4);
  int* condB = (int*)alloc(TST * NB * KW * 4);
  int* lensA = (int*)alloc(NB * KW * 4);
  int* lensB = (int*)alloc(NB * KW * 4);
  int* prefA = (int*)alloc(NB * KW * KW * 4);
  int* prefB = (int*)alloc(NB * KW * KW * 4);

  k_transpose_w<<<DM, 256, 0, stream>>>(W, Wt);
  k_init_state<<<24, 256, 0, stream>>>(probsA, yA, condA, lensA, prefA);

  float *pP = probsA, *pN = probsB;
  int *yP = yA, *yN = yB, *cP = condA, *cN = condB;
  int *lP = lensA, *lN = lensB, *fP = prefA, *fN = prefB;

  for (int t = 0; t < TST; ++t) {
    int Kp = (t == 0) ? 1 : KW;
    int M  = NB * Kp;
    k_embed<<<M, 256, 0, stream>>>(tokE, timE, yP, H, t, Kp);
    int Mrow = ((M >> 4) + MBLK - 1) / MBLK;
    int waves = Mrow * NTI;
    k_gemm<<<(waves + 7) / 8, 256, 0, stream>>>(H, Wt, P, M);
    k_softmax<<<M, 256, 0, stream>>>(P);
    k_topk_update<<<NB, 256, 0, stream>>>(P, pP, yP, cP, lP, fP,
                                          pN, yN, cN, lN, fN, t, Kp);
    { float* tmp = pP; pP = pN; pN = tmp; }
    { int* tmp = yP; yP = yN; yN = tmp; }
    { int* tmp = cP; cP = cN; cN = tmp; }
    { int* tmp = lP; lP = lN; lN = tmp; }
    { int* tmp = fP; fP = fN; fN = tmp; }
  }

  int tot = TST * NB * KW + 2 * NB * KW;
  k_finalize<<<(tot + 255) / 256, 256, 0, stream>>>(cP, lP, pP, d_out);
}
